// ConfidenceGCNConv_77352361001297
// MI455X (gfx1250) — matrix operand order repaired and checked by reading, not compiled, not run on hardware
//
#include <hip/hip_runtime.h>

typedef __attribute__((ext_vector_type(16))) __bf16 v16bf;
typedef __attribute__((ext_vector_type(8)))  __bf16 v8bf;
typedef __attribute__((ext_vector_type(8)))  float  v8f;

#define D_IN   512
#define D_OUT  512
#define TM     128       // block tile rows
#define TN     64        // block tile cols
#define TK     32        // k-step (WMMA bf16 K)
#define LDP    (TK + 8)  // 40 bf16 = 80B row stride: 16B-aligned, breaks 64-bank conflicts

// ---------- helpers ----------
__device__ __forceinline__ unsigned short bf16_bits(float f) {
    unsigned u = __builtin_bit_cast(unsigned, f);
    unsigned r = u + 0x7FFFu + ((u >> 16) & 1u);   // round-to-nearest-even
    return (unsigned short)(r >> 16);
}

// Native CDNA5 no-return f32 atomic add (STOREcnt-tracked; s_endpgm waits idle).
__device__ __forceinline__ void atomic_add_f32(float* p, float v) {
    asm volatile("global_atomic_add_f32 %0, %1, off" : : "v"(p), "v"(v) : "memory");
}

// ---------- K0: zero workspace (agg + deg) ----------
__global__ __launch_bounds__(256) void zero_kernel(float4* p, int n4) {
    int i = blockIdx.x * blockDim.x + threadIdx.x;
    int stride = gridDim.x * blockDim.x;
    float4 z = make_float4(0.f, 0.f, 0.f, 0.f);
    for (; i < n4; i += stride) p[i] = z;
}

// ---------- K1: edge confidence weight + degree count ----------
__global__ __launch_bounds__(256) void edge_kernel(const float* __restrict__ ea,
                                                   const int* __restrict__ row,
                                                   const float* __restrict__ cw,
                                                   const float* __restrict__ cb,
                                                   float* __restrict__ nrm,
                                                   float* __restrict__ deg,
                                                   int E) {
    int e = blockIdx.x * blockDim.x + threadIdx.x;
    if (e >= E) return;
    float z = ea[e * 3 + 0] * cw[0] + ea[e * 3 + 1] * cw[1] + ea[e * 3 + 2] * cw[2] + cb[0];
    nrm[e] = 1.f / (1.f + __expf(-z));   // sigmoid
    atomic_add_f32(&deg[row[e]], 1.0f);
}

// ---------- K2: inverse sqrt degree ----------
__global__ __launch_bounds__(256) void dis_kernel(const float* __restrict__ deg,
                                                  float* __restrict__ dis, int N) {
    int i = blockIdx.x * blockDim.x + threadIdx.x;
    if (i >= N) return;
    float d = deg[i];
    dis[i] = (d > 0.f) ? __frsqrt_rn(d) : 0.f;
}

// ---------- K3: per-edge norm ----------
__global__ __launch_bounds__(256) void norm_kernel(const int* __restrict__ row,
                                                   const int* __restrict__ col,
                                                   const float* __restrict__ dis,
                                                   float* __restrict__ nrm, int E) {
    int e = blockIdx.x * blockDim.x + threadIdx.x;
    if (e >= E) return;
    float nv = dis[row[e]] * dis[col[e]] * nrm[e];
    nrm[e] = (nv != nv) ? 0.f : nv;      // NaN -> 0
}

// ---------- K4: gather-scale-scatter  agg[col] += x[row]*norm ----------
__global__ __launch_bounds__(256) void scatter_kernel(const float* __restrict__ x,
                                                      const int* __restrict__ row,
                                                      const int* __restrict__ col,
                                                      const float* __restrict__ nrm,
                                                      float* __restrict__ agg) {
    int e = blockIdx.x;
    float nv = nrm[e];
    if (nv == 0.f) return;               // uniform across block
    const float* xs = x + (size_t)row[e] * D_IN;
    float* od = agg + (size_t)col[e] * D_IN;
    int d = threadIdx.x;
    atomic_add_f32(&od[d], xs[d] * nv);
    atomic_add_f32(&od[d + 256], xs[d + 256] * nv);
}

// ---------- K5: out = agg @ lin_w.T + lin_b  via v_wmma_f32_16x16x32_bf16 ----------
__global__ __launch_bounds__(256) void gemm_wmma_kernel(const float* __restrict__ A,   // [N, 512] f32
                                                        const float* __restrict__ W,   // [512, 512] f32 (row = out ch)
                                                        const float* __restrict__ bias,
                                                        float* __restrict__ out,       // [N, 512]
                                                        int Nrows) {
    __shared__ alignas(16) __bf16 As[TM][LDP];   // A tile, bf16
    __shared__ alignas(16) __bf16 Bs[TN][LDP];   // Bs[n][k] = W[n0+n][k0+k]
    const int m0   = blockIdx.x * TM;
    const int n0   = blockIdx.y * TN;
    const int tid  = threadIdx.x;
    const int lane = tid & 31;
    const int wave = tid >> 5;

    v8f acc0 = {}, acc1 = {}, acc2 = {}, acc3 = {};

    for (int k0 = 0; k0 < D_IN; k0 += TK) {
        // A tile: 128x32 f32 -> bf16, float4 chunks (8 per row), 1024 chunks / 256 thr = 4 each
        for (int ch = tid; ch < (TM * TK) / 4; ch += 256) {
            int r = ch >> 3, c = (ch & 7) << 2;
            int gr = m0 + r;
            float4 f = (gr < Nrows) ? *(const float4*)&A[(size_t)gr * D_IN + k0 + c]
                                    : make_float4(0.f, 0.f, 0.f, 0.f);
            unsigned lo = ((unsigned)bf16_bits(f.y) << 16) | bf16_bits(f.x);
            unsigned hi = ((unsigned)bf16_bits(f.w) << 16) | bf16_bits(f.z);
            uint2 pk = make_uint2(lo, hi);
            __builtin_memcpy(&As[r][c], &pk, 8);          // ds_store_b64
        }
        // B tile: 64x32, 512 chunks / 256 thr = 2 each (W rows always in range)
        for (int ch = tid; ch < (TN * TK) / 4; ch += 256) {
            int r = ch >> 3, c = (ch & 7) << 2;
            float4 f = *(const float4*)&W[(size_t)(n0 + r) * D_IN + k0 + c];
            unsigned lo = ((unsigned)bf16_bits(f.y) << 16) | bf16_bits(f.x);
            unsigned hi = ((unsigned)bf16_bits(f.w) << 16) | bf16_bits(f.z);
            uint2 pk = make_uint2(lo, hi);
            __builtin_memcpy(&Bs[r][c], &pk, 8);
        }
        __syncthreads();

        // A fragment (ISA 16-bit A 16x32): lanes 0-15 -> K {0..7,16..23}; lanes 16-31 -> K {8..15,24..31}
        const int mrow = (wave << 4) + (lane & 15);
        const int akb  = (lane < 16) ? 0 : 8;
        v8bf alo = *(const v8bf*)&As[mrow][akb];           // ds_load_b128
        v8bf ahi = *(const v8bf*)&As[mrow][akb + 16];      // ds_load_b128
        v16bf afrag = __builtin_shufflevector(alo, ahi,
            0,1,2,3,4,5,6,7,8,9,10,11,12,13,14,15);

        // B fragments (ISA B 32x16): n = lane&15; lanes 0-15 K {0..7,16..23}, lanes 16-31 K {8..15,24..31}
        const int bn  = lane & 15;
        const int bkb = (lane < 16) ? 0 : 8;
#define LOAD_B(rbase)                                                          \
        __builtin_shufflevector(*(const v8bf*)&Bs[(rbase) + bn][bkb],          \
                                *(const v8bf*)&Bs[(rbase) + bn][bkb + 16],     \
                                0,1,2,3,4,5,6,7,8,9,10,11,12,13,14,15)
        v16bf b0 = LOAD_B(0);
        v16bf b1 = LOAD_B(16);
        v16bf b2 = LOAD_B(32);
        v16bf b3 = LOAD_B(48);
#undef LOAD_B
        acc0 = __builtin_amdgcn_wmma_f32_16x16x32_bf16(false, afrag, false, b0, (short)0, acc0, false, false);
        acc1 = __builtin_amdgcn_wmma_f32_16x16x32_bf16(false, afrag, false, b1, (short)0, acc1, false, false);
        acc2 = __builtin_amdgcn_wmma_f32_16x16x32_bf16(false, afrag, false, b2, (short)0, acc2, false, false);
        acc3 = __builtin_amdgcn_wmma_f32_16x16x32_bf16(false, afrag, false, b3, (short)0, acc3, false, false);
        __syncthreads();
    }

    // C/D 16x16 f32 layout: lanes 0-15 -> M = vgpr idx; lanes 16-31 -> M = 8 + vgpr idx; N = lane&15
    const int n_lane = lane & 15;
    const int mbase  = m0 + (wave << 4) + ((lane < 16) ? 0 : 8);
    v8f accs[4] = {acc0, acc1, acc2, acc3};
#pragma unroll
    for (int s = 0; s < 4; ++s) {
        int gn = n0 + s * 16 + n_lane;
        float b = bias[gn];
#pragma unroll
        for (int i = 0; i < 8; ++i) {
            int gm = mbase + i;
            if (gm < Nrows) out[(size_t)gm * D_OUT + gn] = accs[s][i] + b;
        }
    }
}

// ---------- launch ----------
extern "C" void kernel_launch(void* const* d_in, const int* in_sizes, int n_in,
                              void* d_out, int out_size, void* d_ws, size_t ws_size,
                              hipStream_t stream) {
    const float* x    = (const float*)d_in[0];
    const int*   ei   = (const int*)d_in[1];     // [2, E] (row then col)
    const float* ea   = (const float*)d_in[2];   // [E, 3]
    const float* linw = (const float*)d_in[3];   // [512, 512]
    const float* linb = (const float*)d_in[4];   // [512]
    const float* cw   = (const float*)d_in[5];   // [3]
    const float* cb   = (const float*)d_in[6];   // [1]
    float* out = (float*)d_out;

    const int N = in_sizes[0] / D_IN;
    const int E = in_sizes[2] / 3;
    const int* row = ei;
    const int* col = ei + E;

    // workspace layout (all f32)
    float* agg = (float*)d_ws;              // N * 512
    float* deg = agg + (size_t)N * D_IN;    // N
    float* dis = deg + N;                   // N
    float* nrm = dis + N;                   // E

    // K0: zero agg + deg (contiguous; tail spill into dis harmless, dis fully rewritten)
    {
        size_t total = (size_t)N * D_IN + (size_t)N;
        int n4 = (int)((total + 3) / 4);
        zero_kernel<<<1024, 256, 0, stream>>>((float4*)agg, n4);
    }
    // K1: edge confidence + degree
    edge_kernel<<<(E + 255) / 256, 256, 0, stream>>>(ea, row, cw, cb, nrm, deg, E);
    // K2: deg^-1/2
    dis_kernel<<<(N + 255) / 256, 256, 0, stream>>>(deg, dis, N);
    // K3: per-edge norm
    norm_kernel<<<(E + 255) / 256, 256, 0, stream>>>(row, col, dis, nrm, E);
    // K4: scatter-accumulate (one block per edge, 512 floats per edge)
    scatter_kernel<<<E, 256, 0, stream>>>(x, row, col, nrm, agg);
    // K5: WMMA GEMM  out = agg @ W^T + b
    dim3 grid((N + TM - 1) / TM, D_OUT / TN);
    gemm_wmma_kernel<<<grid, 256, 0, stream>>>(agg, linw, linb, out, N);
}
